// PyramidAttention_1116691497484
// MI455X (gfx1250) — matrix operand order repaired and checked by reading, not compiled, not run on hardware
//
#include <hip/hip_runtime.h>

// ---------------------------------------------------------------------------
// PyramidAttention for MI455X (gfx1250)
//   - both big GEMMs (1024x1024x1536 and 1024x3072x1024 per batch) on
//     v_wmma_f32_16x16x32_f16 with fp32 accumulation
//   - LDS tiles filled with GLOBAL_LOAD_ASYNC_TO_LDS_B128 (ASYNCcnt),
//     double-buffered: tile k+1 streams while tile k computes
// ---------------------------------------------------------------------------

typedef __attribute__((ext_vector_type(16))) _Float16 v16h;
typedef __attribute__((ext_vector_type(8)))  float    v8f;

#define Bn     8
#define Cfull  336
#define Chalf  168
#define HW     1024      // 32*32
#define Lkeys  1024
#define KQ_LD  1536      // pad of Chalf*9 = 1512
#define V_LD   3072      // pad of Cfull*9 = 3024

union Frag { v16h v; uint4 q[2]; };

// Async copy of 32 contiguous bytes global -> LDS (2 x b128), no VGPR staging.
// ISA 08_async_tensor: the instruction offset is added to both the LDS and the global address
__device__ __forceinline__ void async_copy_32B(unsigned lds_addr, const _Float16* gptr) {
    asm volatile(
        "global_load_async_to_lds_b128 %0, %1, off\n\t"
        "global_load_async_to_lds_b128 %0, %1, off offset:16"
        :: "v"(lds_addr), "v"(gptr) : "memory");
}

__device__ __forceinline__ unsigned lds_off(const void* p) {
    // flat LDS address: aperture in [63:32], wave-relative offset in [31:0]
    return (unsigned)(uintptr_t)p;
}

// ---------------------------------------------------------------------------
// depthwise 3x3 + BN(eval) + ReLU  ->  ref
// ---------------------------------------------------------------------------
__global__ __launch_bounds__(256)
void dwconv_bn_relu_kernel(const float* __restrict__ x, const float* __restrict__ wg,
                           const float* __restrict__ bng, const float* __restrict__ bnb,
                           const float* __restrict__ bnm, const float* __restrict__ bnv,
                           float* __restrict__ outp) {
    int p = blockIdx.x * 256 + threadIdx.x;   // 0..1023
    int c = blockIdx.y, b = blockIdx.z;
    int h = p >> 5, w = p & 31;
    const float* xp = x + (((size_t)b * Cfull + c) << 10);
    const float* k  = wg + (size_t)c * 9;
    float s = 0.f;
#pragma unroll
    for (int ki = 0; ki < 3; ki++)
#pragma unroll
        for (int kj = 0; kj < 3; kj++) {
            int hh = h + ki - 1, ww = w + kj - 1;
            if ((unsigned)hh < 32u && (unsigned)ww < 32u)
                s = fmaf(k[ki * 3 + kj], xp[(hh << 5) + ww], s);
        }
    float sc = bng[c] * rsqrtf(bnv[c] + 1e-5f);
    float v  = (s - bnm[c]) * sc + bnb[c];
    outp[(((size_t)b * Cfull + c) << 10) + p] = fmaxf(v, 0.f);
}

// ---------------------------------------------------------------------------
// 1x1 conv + PReLU (weight row scalar-broadcast per block)
// ---------------------------------------------------------------------------
__global__ __launch_bounds__(256)
void conv1x1_prelu_kernel(const float* __restrict__ in, const float* __restrict__ W,
                          const float* __restrict__ bias, const float* __restrict__ alpha,
                          float* __restrict__ outp, int Ci, int Co) {
    int p = blockIdx.x * 256 + threadIdx.x;
    int o = blockIdx.y, b = blockIdx.z;
    const float* ip = in + (((size_t)b * Ci) << 10) + p;
    const float* wp = W + (size_t)o * Ci;
    float s = bias[o];
    for (int c = 0; c < Ci; c++) s = fmaf(wp[c], ip[(size_t)c << 10], s);
    float a = alpha[0];
    outp[(((size_t)b * Co + o) << 10) + p] = (s >= 0.f) ? s : a * s;
}

// ---------------------------------------------------------------------------
// Build padded patch matrices (f16). One block per (b, patch l).
// do_norm: L2-normalize (floor 1e-4).  transpose_out: write [j][l] (for V^T).
// ---------------------------------------------------------------------------
__global__ __launch_bounds__(256)
void build_patches_kernel(const float* __restrict__ src, _Float16* __restrict__ dst,
                          int Cch, int kpad, int do_norm, int transpose_out) {
    __shared__ float red[256];
    const int b = blockIdx.y, l = blockIdx.x;
    const int h = l >> 5, w = l & 31;
    const int total = Cch * 9;
    const float* sp = src + (((size_t)b * Cch) << 10);

    float vloc[12];                 // kpad/256 <= 12
    int cnt = 0;
    float ss = 0.f;
    for (int i = threadIdx.x; i < kpad; i += 256) {
        float v = 0.f;
        if (i < total) {
            int c = i / 9, rem = i % 9, ki = rem / 3, kj = rem % 3;
            int hh = h + ki - 1, ww = w + kj - 1;
            if ((unsigned)hh < 32u && (unsigned)ww < 32u)
                v = sp[((size_t)c << 10) + (hh << 5) + ww];
        }
        vloc[cnt++] = v;
        ss += v * v;
    }
    float scale = 1.f;
    if (do_norm) {
        red[threadIdx.x] = ss;
        __syncthreads();
        for (int s = 128; s > 0; s >>= 1) {
            if ((int)threadIdx.x < s) red[threadIdx.x] += red[threadIdx.x + s];
            __syncthreads();
        }
        scale = 1.f / fmaxf(sqrtf(red[0]), 1e-4f);
    }
    cnt = 0;
    if (transpose_out) {
        _Float16* dp = dst + (size_t)b * kpad * 1024 + l;   // [b][j][l]
        for (int i = threadIdx.x; i < kpad; i += 256)
            dp[(size_t)i << 10] = (_Float16)(vloc[cnt++] * scale);
    } else {
        _Float16* dp = dst + ((size_t)b * 1024 + l) * kpad; // [b][l][k]
        for (int i = threadIdx.x; i < kpad; i += 256)
            dp[i] = (_Float16)(vloc[cnt++] * scale);
    }
}

// ---------------------------------------------------------------------------
// Unified f16 WMMA GEMM:  C[m,n] = sum_k A[m,k] * B[n,k]
// block = 256 thr = 8 waves; block tile 128x128; wave tile 32x64 (2x4 WMMA).
// LDS double-buffered, filled via global_load_async_to_lds_b128 (ASYNCcnt).
// ---------------------------------------------------------------------------
__global__ __launch_bounds__(256)
void gemm_f16_kernel(const _Float16* __restrict__ A, const _Float16* __restrict__ B,
                     float* __restrict__ C,
                     int lda, int ldb, int ldc, int Kdim,
                     size_t strideA, size_t strideB, size_t strideC) {
    __shared__ _Float16 As[2][128 * 48];
    __shared__ _Float16 Bs[2][128 * 48];
    const int b = blockIdx.z;
    const int tileM = blockIdx.y * 128;
    const int tileN = blockIdx.x * 128;
    const _Float16* gA = A + (size_t)b * strideA;
    const _Float16* gB = B + (size_t)b * strideB;
    float* gC = C + (size_t)b * strideC;

    const int t = threadIdx.x;
    const int wave = t >> 5, wr = wave >> 1, wc = wave & 1;
    const int lane = t & 31, half = lane >> 4, l15 = lane & 15;
    const int lrow = t >> 1;            // 0..127
    const int hseg = (t & 1) * 16;      // 16 halves = 32B

    const _Float16* gArow = gA + (size_t)(tileM + lrow) * lda + hseg;
    const _Float16* gBrow = gB + (size_t)(tileN + lrow) * ldb + hseg;
    const unsigned ldsA0 = lds_off(&As[0][lrow * 48 + hseg]);
    const unsigned ldsB0 = lds_off(&Bs[0][lrow * 48 + hseg]);
    const unsigned bufstep = lds_off(&As[1][0]) - lds_off(&As[0][0]);

    v8f acc[2][4];
    const v8f vzero = {0.f, 0.f, 0.f, 0.f, 0.f, 0.f, 0.f, 0.f};
#pragma unroll
    for (int mi = 0; mi < 2; mi++)
#pragma unroll
        for (int ni = 0; ni < 4; ni++) acc[mi][ni] = vzero;

    const int nk = Kdim >> 5;
    // prologue: stream tile 0 into buffer 0
    async_copy_32B(ldsA0, gArow);
    async_copy_32B(ldsB0, gBrow);

    for (int it = 0; it < nk; ++it) {
        const int buf = it & 1;
        if (it + 1 < nk) {
            // stream next tile into the other buffer (protected by trailing
            // barrier of iteration it-1), then wait for the current tile's
            // 4 async ops (in-order completion): ASYNCcnt <= 4
            async_copy_32B(ldsA0 + (buf ^ 1) * bufstep, gArow + (it + 1) * 32);
            async_copy_32B(ldsB0 + (buf ^ 1) * bufstep, gBrow + (it + 1) * 32);
            if (it + 2 < nk) {
                __builtin_prefetch(gArow + (it + 2) * 32, 0, 1);
                __builtin_prefetch(gBrow + (it + 2) * 32, 0, 1);
            }
            asm volatile("s_wait_asynccnt 0x4" ::: "memory");
        } else {
            asm volatile("s_wait_asynccnt 0x0" ::: "memory");
        }
        __syncthreads();

        Frag afrag[2], bfrag[4];
#pragma unroll
        for (int mi = 0; mi < 2; mi++) {
            // A layout: lane&15 = M row; lanes<16: e0-7->K0-7, e8-15->K16-23;
            //           lanes>=16: e0-7->K8-15, e8-15->K24-31
            const uint4* rp = (const uint4*)&As[buf][(wr * 32 + mi * 16 + l15) * 48];
            afrag[mi].q[0] = rp[half];
            afrag[mi].q[1] = rp[2 + half];
        }
#pragma unroll
        for (int ni = 0; ni < 4; ni++) {
            // B layout: lane&15 = N col; lanes<16: e0-7->K0-7, e8-15->K16-23;
            //           lanes>=16: e0-7->K8-15, e8-15->K24-31
            const uint4* cp = (const uint4*)&Bs[buf][(wc * 64 + ni * 16 + l15) * 48];
            bfrag[ni].q[0] = cp[half];
            bfrag[ni].q[1] = cp[2 + half];
        }
#pragma unroll
        for (int mi = 0; mi < 2; mi++)
#pragma unroll
            for (int ni = 0; ni < 4; ni++)
                acc[mi][ni] = __builtin_amdgcn_wmma_f32_16x16x32_f16(
                    false, afrag[mi].v, false, bfrag[ni].v,
                    (short)0, acc[mi][ni], false, false);
        __syncthreads();
    }

#pragma unroll
    for (int mi = 0; mi < 2; mi++)
#pragma unroll
        for (int ni = 0; ni < 4; ni++)
#pragma unroll
            for (int r = 0; r < 8; r++) {
                int M = tileM + wr * 32 + mi * 16 + half * 8 + r;
                int N = tileN + wc * 64 + ni * 16 + l15;
                gC[(size_t)M * ldc + N] = acc[mi][ni][r];
            }
}

// ---------------------------------------------------------------------------
// Column softmax over keys l; emits A^T in f16: At[b][q][l]
// (reads are fully coalesced: one thread per query column)
// ---------------------------------------------------------------------------
__global__ __launch_bounds__(256)
void softmax_kernel(const float* __restrict__ S, _Float16* __restrict__ At) {
    int b = blockIdx.y;
    int q = blockIdx.x * 256 + threadIdx.x;
    const float* col = S + ((size_t)b << 20) + q;
    float m = -3.4e38f;
    for (int l = 0; l < 1024; l++) m = fmaxf(m, col[(size_t)l << 10] * 10.f);
    float s = 0.f;
    for (int l = 0; l < 1024; l++) s += __expf(col[(size_t)l << 10] * 10.f - m);
    float inv = 1.f / s;
    _Float16* row = At + ((size_t)b << 20) + ((size_t)q << 10);
    for (int l = 0; l < 1024; l++)
        row[l] = (_Float16)(__expf(col[(size_t)l << 10] * 10.f - m) * inv);
}

// ---------------------------------------------------------------------------
// Overlap-add gather + residual (deterministic, no atomics):
// out = 5*x + 0.25 * sum_{ki,kj} T[q(h,w,ki,kj)][c*9+ki*3+kj]
// ---------------------------------------------------------------------------
__global__ __launch_bounds__(256)
void gather_out_kernel(const float* __restrict__ x, const float* __restrict__ T,
                       float* __restrict__ outp) {
    int p = blockIdx.x * 256 + threadIdx.x;
    int c = blockIdx.y, b = blockIdx.z;
    int h = p >> 5, w = p & 31;
    const float* gT = T + (size_t)b * Lkeys * V_LD;
    float acc = 0.f;
#pragma unroll
    for (int ki = 0; ki < 3; ki++)
#pragma unroll
        for (int kj = 0; kj < 3; kj++) {
            int qh = h - ki + 1, qw = w - kj + 1;
            if ((unsigned)qh < 32u && (unsigned)qw < 32u) {
                int q = (qh << 5) + qw;
                acc += gT[(size_t)q * V_LD + c * 9 + ki * 3 + kj];
            }
        }
    size_t idx = (((size_t)b * Cfull + c) << 10) + p;
    outp[idx] = 5.f * x[idx] + 0.25f * acc;
}

// ---------------------------------------------------------------------------
extern "C" void kernel_launch(void* const* d_in, const int* in_sizes, int n_in,
                              void* d_out, int out_size, void* d_ws, size_t ws_size,
                              hipStream_t stream) {
    const float* x    = (const float*)d_in[0];
    const float* w_mb = (const float*)d_in[1];
    const float* b_mb = (const float*)d_in[2];
    const float* a_mb = (const float*)d_in[3];
    const float* w_m  = (const float*)d_in[4];
    const float* b_m  = (const float*)d_in[5];
    const float* a_m  = (const float*)d_in[6];
    const float* w_as = (const float*)d_in[7];
    const float* b_as = (const float*)d_in[8];
    const float* a_as = (const float*)d_in[9];
    const float* w_g  = (const float*)d_in[10];
    const float* bn_g = (const float*)d_in[11];
    const float* bn_b = (const float*)d_in[12];
    const float* bn_m = (const float*)d_in[13];
    const float* bn_v = (const float*)d_in[14];
    float* outp = (float*)d_out;
    (void)in_sizes; (void)n_in; (void)out_size; (void)ws_size;

    char* ws = (char*)d_ws;
    size_t off = 0;
    auto take = [&](size_t bytes) -> char* {
        char* p = ws + off;
        off = (off + bytes + 255) & ~(size_t)255;
        return p;
    };
    float*    ref  = (float*)take((size_t)Bn * Cfull * HW * 4);         // 11.0 MB
    float*    mb   = (float*)take((size_t)Bn * Chalf * HW * 4);         //  5.5 MB
    float*    base = (float*)take((size_t)Bn * Cfull * HW * 4);         // 11.0 MB
    float*    refi = (float*)take((size_t)Bn * Chalf * HW * 4);         //  5.5 MB
    _Float16* Qm   = (_Float16*)take((size_t)Bn * Lkeys * KQ_LD * 2);   // 25.2 MB [b][q][k]
    _Float16* Km   = (_Float16*)take((size_t)Bn * Lkeys * KQ_LD * 2);   // 25.2 MB [b][l][k]
    _Float16* Vt   = (_Float16*)take((size_t)Bn * V_LD * Lkeys * 2);    // 50.3 MB [b][j][l]
    float*    Smat = (float*)take((size_t)Bn * Lkeys * Lkeys * 4);      // 33.6 MB [b][l][q]
    _Float16* At   = (_Float16*)take((size_t)Bn * Lkeys * Lkeys * 2);   // 16.8 MB [b][q][l]
    float*    Tmat = (float*)take((size_t)Bn * Lkeys * V_LD * 4);       // 100.7 MB [b][q][j]

    // 1) ref = relu(BN(dwconv3x3(x)))
    dwconv_bn_relu_kernel<<<dim3(4, Cfull, Bn), 256, 0, stream>>>(
        x, w_g, bn_g, bn_b, bn_m, bn_v, ref);
    // 2) three 1x1 convs + PReLU
    conv1x1_prelu_kernel<<<dim3(4, Chalf, Bn), 256, 0, stream>>>(x,   w_mb, b_mb, a_mb, mb,   Cfull, Chalf);
    conv1x1_prelu_kernel<<<dim3(4, Cfull, Bn), 256, 0, stream>>>(ref, w_as, b_as, a_as, base, Cfull, Cfull);
    conv1x1_prelu_kernel<<<dim3(4, Chalf, Bn), 256, 0, stream>>>(ref, w_m,  b_m,  a_m,  refi, Cfull, Chalf);
    // 3) padded patch matrices (f16); V is written transposed [j][l]
    build_patches_kernel<<<dim3(1024, Bn), 256, 0, stream>>>(mb,   Qm, Chalf, KQ_LD, 0, 0);
    build_patches_kernel<<<dim3(1024, Bn), 256, 0, stream>>>(refi, Km, Chalf, KQ_LD, 1, 0);
    build_patches_kernel<<<dim3(1024, Bn), 256, 0, stream>>>(base, Vt, Cfull, V_LD,  0, 1);
    // 4) S[l][q] = sum_k K[l,k] Q[q,k]
    gemm_f16_kernel<<<dim3(8, 8, Bn), 256, 0, stream>>>(
        Km, Qm, Smat, KQ_LD, KQ_LD, 1024, KQ_LD,
        (size_t)Lkeys * KQ_LD, (size_t)Lkeys * KQ_LD, (size_t)1 << 20);
    // 5) softmax over keys per query column -> At[q][l] (f16)
    softmax_kernel<<<dim3(4, Bn), 256, 0, stream>>>(Smat, At);
    // 6) T[q][j] = sum_l At[q,l] Vt[j,l]
    gemm_f16_kernel<<<dim3(24, 8, Bn), 256, 0, stream>>>(
        At, Vt, Tmat, 1024, 1024, V_LD, 1024,
        (size_t)1 << 20, (size_t)V_LD * Lkeys, (size_t)Lkeys * V_LD);
    // 7) out = 5*x + 0.25 * overlap_add(T)
    gather_out_kernel<<<dim3(4, Cfull, Bn), 256, 0, stream>>>(x, Tmat, outp);
}
